// GraphConvolution_18270790877922
// MI455X (gfx1250) — matrix operand order repaired and checked by reading, not compiled, not run on hardware
//
#include <hip/hip_runtime.h>
#include <hip/hip_fp16.h>

typedef __attribute__((ext_vector_type(16))) _Float16 v16h;
typedef __attribute__((ext_vector_type(8)))  _Float16 v8h;
typedef __attribute__((ext_vector_type(4)))  _Float16 v4h;
typedef __attribute__((ext_vector_type(8)))  float    v8f;

#define NNODES 100000
#define NEDGES 1600000
#define DF     64
#define ALPHA  0.1f
#define LAMDA  0.5f
#define LDW    72   // padded LDS row stride (f16 elems): 144B = 36 banks, gcd(36,64)=4

// ---------------------------------------------------------------------------
// Kernel 1: zero the hi accumulator (lives in d_ws). N*D/4 float4 = 1.6M
// ---------------------------------------------------------------------------
__global__ __launch_bounds__(256) void hi_zero(float4* __restrict__ hi) {
    const float4 z = {0.f, 0.f, 0.f, 0.f};
    hi[(size_t)blockIdx.x * 256 + threadIdx.x] = z;
}

// ---------------------------------------------------------------------------
// Kernel 2: SpMM scatter. One wave32 per edge: 32 lanes x float2 covers the
// 256B source row coalesced; scatter-add lands as native f32 atomics in L2
// (x and hi are both L2-resident: 25.6 MB each vs 192 MB L2).
// ---------------------------------------------------------------------------
__global__ __launch_bounds__(256) void spmm_scatter(
    const float* __restrict__ x, const int* __restrict__ src,
    const int* __restrict__ dst, const float* __restrict__ ew,
    float* __restrict__ hi) {
    const int e    = (int)((blockIdx.x * 256u + threadIdx.x) >> 5);
    const int lane = threadIdx.x & 31;
    if (e >= NEDGES) return;

    const int   s = src[e];
    const int   d = dst[e];
    const float w = ew[e];

    // speculative prefetch of a source row ~256 edges ahead (global_prefetch_b8)
    if (e + 256 < NEDGES) {
        const int sp = src[e + 256];
        __builtin_prefetch(x + (size_t)sp * DF + lane * 2, 0, 0);
    }

    const float2 v = *reinterpret_cast<const float2*>(x + (size_t)s * DF + lane * 2);
    float* h = hi + (size_t)d * DF + lane * 2;
    unsafeAtomicAdd(h + 0, v.x * w);
    unsafeAtomicAdd(h + 1, v.y * w);
}

// ---------------------------------------------------------------------------
// Kernel 3: fused support + GEMM (WMMA f16 -> f32 acc) + theta blend + residual.
// 256 threads = 8 waves; each wave owns one 16-row tile, full 64-col output.
// All fragments are preloaded (2x ds_load_b128 each), then the 8 WMMAs issue
// back-to-back behind a single DS wait.
// ---------------------------------------------------------------------------
__global__ __launch_bounds__(256) void gcn_fused_gemm(
    const float* __restrict__ x, const float* __restrict__ h0,
    const float* __restrict__ hi, const float* __restrict__ W,
    const int* __restrict__ lptr, float* __restrict__ out) {

    // W stored TRANSPOSED: wt[n*LDW + k]. B fragment (lane l, tile ks,nt) is
    // two contiguous 16B runs in row nt*16+mn at k = ks*32 + g*8 and ks*32 + 16 + g*8.
    __shared__ alignas(32) _Float16 wt[DF * LDW];
    // support tile, row-major: st[m*LDW + k]. A fragment is two contiguous
    // 16B runs at k = ks*32 + g*8 and k = ks*32 + 16 + g*8.
    __shared__ alignas(32) _Float16 slds[8][16 * LDW];

    // cooperative load + transpose + f16-convert of the 64x64 weight
    for (int i = threadIdx.x; i < DF * DF; i += 256) {
        const int k = i >> 6, n = i & 63;      // W is [k][n] row-major in memory
        wt[n * LDW + k] = (_Float16)W[i];
    }
    __syncthreads();

    const int wave = threadIdx.x >> 5;
    const int lane = threadIdx.x & 31;
    const int g    = lane >> 4;        // lane half-group (CDNA5 wave32 layout)
    const int mn   = lane & 15;
    const int tile = blockIdx.x * 8 + wave;
    if (tile >= (NNODES / 16)) return;         // wave-uniform; after the barrier
    const int rowBase = tile * 16;

    // stage support = 0.9*hi + 0.1*h0 for this wave's 16x64 tile (f16, LDS).
    // float4 per lane, two rows per step (group g picks the row). Same-wave DS
    // ops are in-order, so no barrier is needed before the fragment reads.
    _Float16* st = &slds[wave][0];
    #pragma unroll
    for (int it = 0; it < 8; ++it) {
        const int r = it * 2 + g;
        const int c = mn * 4;
        const size_t row = (size_t)(rowBase + r) * DF + c;
        const float4 a = *reinterpret_cast<const float4*>(hi + row);
        const float4 b = *reinterpret_cast<const float4*>(h0 + row);
        v4h p;
        p.x = (_Float16)((1.0f - ALPHA) * a.x + ALPHA * b.x);
        p.y = (_Float16)((1.0f - ALPHA) * a.y + ALPHA * b.y);
        p.z = (_Float16)((1.0f - ALPHA) * a.z + ALPHA * b.z);
        p.w = (_Float16)((1.0f - ALPHA) * a.w + ALPHA * b.w);
        *reinterpret_cast<v4h*>(st + r * LDW + c) = p;          // ds_store_b64
    }

    // ---- preload ALL fragments, then issue the 8 WMMAs back-to-back --------
    v16h afrag[2];
    v16h bfrag[2][4];
    #pragma unroll
    for (int ks = 0; ks < 2; ++ks) {
        // A fragment, 16x32 f16 (ISA 7.12.2): row M = mn for both lane groups;
        // group g covers K {g*8..g*8+7} and {16+g*8..16+g*8+7} within the step.
        const v8h alo = *reinterpret_cast<const v8h*>(st + mn * LDW + ks * 32 + g * 8);
        const v8h ahi = *reinterpret_cast<const v8h*>(st + mn * LDW + ks * 32 + 16 + g * 8);
        afrag[ks] = __builtin_shufflevector(alo, ahi,
            0, 1, 2, 3, 4, 5, 6, 7, 8, 9, 10, 11, 12, 13, 14, 15);
        #pragma unroll
        for (int nt = 0; nt < 4; ++nt) {
            // B fragment, 32x16 f16: lane needs K = ks*32 + g*8 + (0..7) and
            // ks*32 + 16 + g*8 + (0..7) of column n = nt*16 + mn in transposed wt.
            const _Float16* bp = wt + (nt * 16 + mn) * LDW + ks * 32 + g * 8;
            const v8h blo = *reinterpret_cast<const v8h*>(bp);
            const v8h bhi = *reinterpret_cast<const v8h*>(bp + 16);
            bfrag[ks][nt] = __builtin_shufflevector(blo, bhi,
                0, 1, 2, 3, 4, 5, 6, 7, 8, 9, 10, 11, 12, 13, 14, 15);
        }
    }

    v8f acc[4] = {};
    #pragma unroll
    for (int ks = 0; ks < 2; ++ks) {
        #pragma unroll
        for (int nt = 0; nt < 4; ++nt) {
            acc[nt] = __builtin_amdgcn_wmma_f32_16x16x32_f16(
                false, afrag[ks], false, bfrag[ks][nt], (short)0, acc[nt],
                false, false);
        }
    }

    // epilogue: out = theta*(S@W) + (1-theta)*S + x.  Re-derive S from global
    // hi/h0 (L2-resident, coalesced) so the linear term keeps f32 precision.
    // Single v_rcp_f32 instead of a software-correct f32 divide.
    const float theta = __logf(LAMDA * __builtin_amdgcn_rcpf((float)(*lptr)) + 1.0f);
    const float omt   = 1.0f - theta;
    #pragma unroll
    for (int nt = 0; nt < 4; ++nt) {
        #pragma unroll
        for (int v = 0; v < 8; ++v) {
            const int r   = g * 8 + v;              // C/D layout: lanes 16-31 -> M+8
            const int col = nt * 16 + mn;
            const size_t idx = (size_t)(rowBase + r) * DF + col;
            const float s = (1.0f - ALPHA) * hi[idx] + ALPHA * h0[idx];
            out[idx] = theta * acc[nt][v] + omt * s + x[idx];
        }
    }
}

// ---------------------------------------------------------------------------
extern "C" void kernel_launch(void* const* d_in, const int* in_sizes, int n_in,
                              void* d_out, int out_size, void* d_ws, size_t ws_size,
                              hipStream_t stream) {
    const float* x    = (const float*)d_in[0];
    const float* h0   = (const float*)d_in[1];
    const int*   esrc = (const int*)d_in[2];
    const int*   edst = (const int*)d_in[3];
    const float* ew   = (const float*)d_in[4];
    const float* W    = (const float*)d_in[5];
    const int*   lp   = (const int*)d_in[6];
    float* out = (float*)d_out;
    float* hi  = (float*)d_ws;   // N*D*4 = 25.6 MB scratch accumulator

    hi_zero<<<(NNODES * DF / 4) / 256, 256, 0, stream>>>((float4*)hi);

    const int scatter_blocks = (int)(((size_t)NEDGES * 32 + 255) / 256); // 200000
    spmm_scatter<<<scatter_blocks, 256, 0, stream>>>(x, esrc, edst, ew, hi);

    const int tiles = NNODES / 16;               // 6250 (N divisible by 16)
    gcn_fused_gemm<<<(tiles + 7) / 8, 256, 0, stream>>>(x, h0, hi, W, lp, out);
}
